// AutoregressiveCDF_45346264711670
// MI455X (gfx1250) — matrix operand order repaired and checked by reading, not compiled, not run on hardware
//
#include <hip/hip_runtime.h>
#include <hip/hip_bf16.h>

// ---------------------------------------------------------------------------
// Problem constants
// ---------------------------------------------------------------------------
#define FEAT   64
#define HID    512
#define CTX    512
#define NBINS  30
#define MULTP  91          // 3*NBINS+1
#define NPADF  96          // per-feature padded output width (6 tiles of 16)
#define NOUTP  (64 * 96)   // 6144 padded output columns
#define ROWS   32          // batch rows per workgroup (2 M-tiles)

// swizzled-weight offsets in halves (bf16) inside d_ws
#define OFF_WIN   0L                    // 64x512      -> 32768
#define OFF_WCIN  32768L                // 512x512     -> 262144
#define OFF_WB1   294912L               // 3x512x512
#define OFF_WB2   1081344L
#define OFF_WCB   1867776L
#define OFF_WOUT  2654208L              // 512x6144    -> 3145728
#define OFF_BOUT_BYTES 11599872L        // padded b_out (6144 floats) after weights

typedef __attribute__((ext_vector_type(16))) __bf16 v16bf;
typedef __attribute__((ext_vector_type(8)))  float  v8f;

__device__ __forceinline__ unsigned short f2bf(float x) {
    union { float f; unsigned u; } v; v.f = x;
    unsigned r = v.u + 0x7FFFu + ((v.u >> 16) & 1u);   // round-to-nearest-even
    return (unsigned short)(r >> 16);
}

union FragB16 { v16bf v; unsigned u[8]; uint4 q[2]; };

// A fragment (16x32 bf16, ISA layout) from an LDS tile at row offset m0.
__device__ __forceinline__ v16bf load_a_lds(const unsigned short* lds, int strideH,
                                            int k0, int m0) {
    const int lane = threadIdx.x & 31;
    const int m = m0 + (lane & 15);
    const int kh = (lane >> 4) * 8;
    const unsigned* p = reinterpret_cast<const unsigned*>(lds + m * strideH + k0 + kh);
    FragB16 f;
    f.u[0] = p[0];  f.u[1] = p[1];  f.u[2] = p[2];  f.u[3] = p[3];
    f.u[4] = p[8];  f.u[5] = p[9];  f.u[6] = p[10]; f.u[7] = p[11];
    return f.v;
}

// B fragment (32x16 bf16) from pre-swizzled global weights (L2-resident).
__device__ __forceinline__ v16bf load_b_glb(const unsigned short* gsw, long blk) {
    const int lane = threadIdx.x & 31;
    const uint4* p = reinterpret_cast<const uint4*>(gsw + blk * 512 + (long)lane * 16);
    FragB16 f;
    f.q[0] = p[0];
    f.q[1] = p[1];
    return f.v;
}

__device__ __forceinline__ float softplus_f(float x) {
    return (x > 20.f) ? x : log1pf(__expf(x));
}

#define WMMA_BF16(A, B, Cacc) \
    __builtin_amdgcn_wmma_f32_16x16x32_bf16(false, (A), false, (B), (short)0, (Cacc), false, false)

// ---------------------------------------------------------------------------
// Weight preparation: apply MADE masks analytically, convert to bf16, swizzle
// into WMMA fragment order. mode: 0=plain, 1=input mask, 2=hidden mask,
// 3=output mask with 91->96 column padding.
// ---------------------------------------------------------------------------
__global__ void prep_weights(const float* __restrict__ src, unsigned short* __restrict__ dst,
                             int K, int Nsrc, int Npad, int mode) {
    const int total = K * Npad;
    for (int idx = blockIdx.x * blockDim.x + threadIdx.x; idx < total;
         idx += gridDim.x * blockDim.x) {
        const int blk  = idx >> 9;          // 512 halves per fragment block
        const int r    = idx & 511;
        const int lane = r >> 4;
        const int i    = r & 15;
        const int ktiles = K >> 5;
        const int nt = blk / ktiles;
        const int kt = blk - nt * ktiles;
        const int n = nt * 16 + (lane & 15);
        const int k = kt * 32 + (lane >> 4) * 8 + (i & 7) + (i >> 3) * 16;
        float v = 0.f;
        if (mode == 3) {
            const int f = n / NPADF;
            const int c = n - f * NPADF;
            if (c < MULTP) {
                const int dh = k % 63 + 1;
                if (f + 1 > dh) v = src[(long)k * Nsrc + f * MULTP + c];
            }
        } else {
            const float w = src[(long)k * Nsrc + n];
            if (mode == 0)      v = w;
            else if (mode == 1) v = ((n % 63 + 1) >= (k + 1)) ? w : 0.f;       // d_h[n] >= d_in[k]
            else                v = ((n % 63 + 1) >= (k % 63 + 1)) ? w : 0.f;  // d_h[n] >= d_h[k]
        }
        dst[idx] = f2bf(v);
    }
}

__global__ void prep_bout(const float* __restrict__ b_out, float* __restrict__ dst) {
    const int idx = blockIdx.x * blockDim.x + threadIdx.x;
    if (idx < NOUTP) {
        const int f = idx / NPADF, c = idx - (idx / NPADF) * NPADF;
        dst[idx] = (c < MULTP) ? b_out[f * MULTP + c] : 0.f;
    }
}

// ---------------------------------------------------------------------------
// Fused MADE + RQ-spline kernel: 32 batch rows per block, 8 waves.
// Each wave owns 4 N-tiles and BOTH 16-row M-tiles, so every B fragment feeds
// two WMMAs.  K-loops are ping-pong double-buffered (kt += 2): each fragment
// slot is refilled directly by loads (never by register copies), keeping one
// slot of load lookahead without mov-rotation overhead.
// LDS layout (160 KB total -> 2 workgroups per WGP):
//   [0      , 65536 ) t   fp32 [32][512]   (aliased by pbuf in output stage)
//   [65536  , 98304 ) hB  bf16 [32][512]   (pred tile / h2; aliased by pbuf)
//   [98304  , 131072) hC  bf16 [32][512]   (relu(t) / t_bf16)
//   [131072 , 163840) ctxB bf16 [32][512]  (aliased by parts at the end)
// ---------------------------------------------------------------------------
__global__ __launch_bounds__(256)
void made_spline_kernel(const float* __restrict__ pred,
                        const float* __restrict__ ctx,
                        const float* __restrict__ b_in,
                        const float* __restrict__ bc_in,
                        const float* __restrict__ bb1,
                        const float* __restrict__ bb2,
                        const float* __restrict__ bcb,
                        const unsigned short* __restrict__ wsw,
                        const float* __restrict__ boutp,
                        float* __restrict__ out) {
    __shared__ unsigned char smem[163840] __attribute__((aligned(16)));
    float*          tT    = (float*)smem;                      // [32][512] fp32 residual
    unsigned short* hB    = (unsigned short*)(smem + 65536);   // bf16 (pred tile / h2)
    unsigned short* hC    = (unsigned short*)(smem + 98304);   // bf16 (relu(t) / t_bf16)
    unsigned short* ctxB  = (unsigned short*)(smem + 131072);  // [32][512] bf16 context
    float*          pbuf  = (float*)smem;                      // out stage: 8 x [32][96] fp32 (96 KB)
    float*          parts = (float*)(smem + 131072);           // [8][32] product partials

    const int tid  = threadIdx.x;
    const int wave = tid >> 5;
    const int lane = tid & 31;
    const int mhi  = (lane >> 4) * 8;
    const int row0 = blockIdx.x * ROWS;

    // stage context (bf16) and predicates (bf16, stride 64 in hB)
    #pragma unroll 4
    for (int j = 0; j < 64; ++j) {
        int idx = tid + j * 256;                 // 32*512
        int r = idx >> 9, c = idx & 511;
        ctxB[r * HID + c] = f2bf(ctx[(long)(row0 + r) * CTX + c]);
    }
    #pragma unroll
    for (int j = 0; j < 8; ++j) {
        int idx = tid + j * 256;                 // 32*64
        int r = idx >> 6, c = idx & 63;
        hB[r * 64 + c] = f2bf(pred[(long)(row0 + r) * FEAT + c]);
    }
    __syncthreads();

    // ---- input layer: t = pred@Win_m + ctx@Wcin + b_in + bc_in ------------
    for (int q = 0; q < 4; ++q) {
        const int nt = wave * 4 + q;
        v8f acc0 = {}, acc1 = {};
        // predicate part (K=64, 2 steps, small: no pipelining needed)
        #pragma unroll
        for (int kt = 0; kt < 2; ++kt) {
            v16bf b  = load_b_glb(wsw + OFF_WIN, (long)nt * 2 + kt);
            v16bf a0 = load_a_lds(hB, 64, kt * 32, 0);
            v16bf a1 = load_a_lds(hB, 64, kt * 32, 16);
            acc0 = WMMA_BF16(a0, b, acc0);
            acc1 = WMMA_BF16(a1, b, acc1);
        }
        // context part (K=512, 16 steps, ping-pong double-buffered)
        {
            const long base = (long)nt * 16;
            v16bf bA  = load_b_glb(wsw + OFF_WCIN, base);
            v16bf a0A = load_a_lds(ctxB, HID, 0, 0);
            v16bf a1A = load_a_lds(ctxB, HID, 0, 16);
            #pragma unroll 2
            for (int kt = 0; kt < 16; kt += 2) {
                v16bf bB  = load_b_glb(wsw + OFF_WCIN, base + kt + 1);
                v16bf a0B = load_a_lds(ctxB, HID, (kt + 1) * 32, 0);
                v16bf a1B = load_a_lds(ctxB, HID, (kt + 1) * 32, 16);
                acc0 = WMMA_BF16(a0A, bA, acc0);
                acc1 = WMMA_BF16(a1A, bA, acc1);
                const int k2 = (kt + 2) & 15;
                bA  = load_b_glb(wsw + OFF_WCIN, base + k2);
                a0A = load_a_lds(ctxB, HID, k2 * 32, 0);
                a1A = load_a_lds(ctxB, HID, k2 * 32, 16);
                acc0 = WMMA_BF16(a0B, bB, acc0);
                acc1 = WMMA_BF16(a1B, bB, acc1);
            }
        }
        const int n = nt * 16 + (lane & 15);
        const float bias = b_in[n] + bc_in[n];
        #pragma unroll
        for (int r = 0; r < 8; ++r) {
            tT[(r + mhi) * HID + n]      = acc0[r] + bias;
            tT[(16 + r + mhi) * HID + n] = acc1[r] + bias;
        }
    }
    __syncthreads();

    // ---- 3 residual blocks ------------------------------------------------
    for (int blk = 0; blk < 3; ++blk) {
        #pragma unroll 4
        for (int j = 0; j < 64; ++j) {           // h1 = relu(t) -> bf16
            int idx = tid + j * 256;
            float v = tT[idx];
            hC[idx] = f2bf(v > 0.f ? v : 0.f);
        }
        __syncthreads();

        const unsigned short* W1 = wsw + OFF_WB1 + (long)blk * 262144;
        for (int q = 0; q < 4; ++q) {            // h2 = relu(h1@W1 + b1)
            const int nt = wave * 4 + q;
            const long base = (long)nt * 16;
            v8f acc0 = {}, acc1 = {};
            v16bf bA  = load_b_glb(W1, base);
            v16bf a0A = load_a_lds(hC, HID, 0, 0);
            v16bf a1A = load_a_lds(hC, HID, 0, 16);
            #pragma unroll 2
            for (int kt = 0; kt < 16; kt += 2) {
                v16bf bB  = load_b_glb(W1, base + kt + 1);
                v16bf a0B = load_a_lds(hC, HID, (kt + 1) * 32, 0);
                v16bf a1B = load_a_lds(hC, HID, (kt + 1) * 32, 16);
                acc0 = WMMA_BF16(a0A, bA, acc0);
                acc1 = WMMA_BF16(a1A, bA, acc1);
                const int k2 = (kt + 2) & 15;
                bA  = load_b_glb(W1, base + k2);
                a0A = load_a_lds(hC, HID, k2 * 32, 0);
                a1A = load_a_lds(hC, HID, k2 * 32, 16);
                acc0 = WMMA_BF16(a0B, bB, acc0);
                acc1 = WMMA_BF16(a1B, bB, acc1);
            }
            const int n = nt * 16 + (lane & 15);
            const float bias = bb1[blk * HID + n];
            #pragma unroll
            for (int r = 0; r < 8; ++r) {
                float v0 = acc0[r] + bias;
                float v1 = acc1[r] + bias;
                hB[(r + mhi) * HID + n]      = f2bf(v0 > 0.f ? v0 : 0.f);
                hB[(16 + r + mhi) * HID + n] = f2bf(v1 > 0.f ? v1 : 0.f);
            }
        }
        __syncthreads();

        const unsigned short* W2 = wsw + OFF_WB2 + (long)blk * 262144;
        const unsigned short* Wg = wsw + OFF_WCB + (long)blk * 262144;
        for (int q = 0; q < 4; ++q) {            // t += (h2@W2 + b2) * sigmoid(ctx@Wg + bg)
            const int nt = wave * 4 + q;
            const long base = (long)nt * 16;
            v8f a20 = {}, a21 = {}, ag0 = {}, ag1 = {};
            v16bf b2A = load_b_glb(W2, base);
            v16bf bgA = load_b_glb(Wg, base);
            v16bf h0A = load_a_lds(hB, HID, 0, 0);
            v16bf h1A = load_a_lds(hB, HID, 0, 16);
            v16bf c0A = load_a_lds(ctxB, HID, 0, 0);
            v16bf c1A = load_a_lds(ctxB, HID, 0, 16);
            for (int kt = 0; kt < 16; kt += 2) {
                v16bf b2B = load_b_glb(W2, base + kt + 1);
                v16bf bgB = load_b_glb(Wg, base + kt + 1);
                v16bf h0B = load_a_lds(hB, HID, (kt + 1) * 32, 0);
                v16bf h1B = load_a_lds(hB, HID, (kt + 1) * 32, 16);
                v16bf c0B = load_a_lds(ctxB, HID, (kt + 1) * 32, 0);
                v16bf c1B = load_a_lds(ctxB, HID, (kt + 1) * 32, 16);
                a20 = WMMA_BF16(h0A, b2A, a20);
                a21 = WMMA_BF16(h1A, b2A, a21);
                ag0 = WMMA_BF16(c0A, bgA, ag0);
                ag1 = WMMA_BF16(c1A, bgA, ag1);
                const int k2 = (kt + 2) & 15;
                b2A = load_b_glb(W2, base + k2);
                bgA = load_b_glb(Wg, base + k2);
                h0A = load_a_lds(hB, HID, k2 * 32, 0);
                h1A = load_a_lds(hB, HID, k2 * 32, 16);
                c0A = load_a_lds(ctxB, HID, k2 * 32, 0);
                c1A = load_a_lds(ctxB, HID, k2 * 32, 16);
                a20 = WMMA_BF16(h0B, b2B, a20);
                a21 = WMMA_BF16(h1B, b2B, a21);
                ag0 = WMMA_BF16(c0B, bgB, ag0);
                ag1 = WMMA_BF16(c1B, bgB, ag1);
            }
            const int n = nt * 16 + (lane & 15);
            const float b2v = bb2[blk * HID + n];
            const float bgv = bcb[blk * HID + n];
            #pragma unroll
            for (int r = 0; r < 8; ++r) {
                float g0 = (a20[r] + b2v) * (1.f / (1.f + __expf(-(ag0[r] + bgv))));
                float g1 = (a21[r] + b2v) * (1.f / (1.f + __expf(-(ag1[r] + bgv))));
                tT[(r + mhi) * HID + n]      += g0;
                tT[(16 + r + mhi) * HID + n] += g1;
            }
        }
        __syncthreads();
    }

    // ---- output layer + spline, one feature at a time per wave ------------
    #pragma unroll 4
    for (int j = 0; j < 64; ++j) {               // t -> bf16 (into hC)
        int idx = tid + j * 256;
        hC[idx] = f2bf(tT[idx]);
    }
    __syncthreads();

    const float c1 = 1.0f - 1e-3f * (float)NBINS;
    const float pscale = 0.044194173824159216f;  // 1/sqrt(512)
    float prod = 1.0f;
    float* pw = pbuf + wave * (ROWS * NPADF);    // [32][96] per wave

    for (int ff = 0; ff < 8; ++ff) {
        const int f = wave * 8 + ff;
        for (int nt6 = 0; nt6 < 6; ++nt6) {
            const long base = ((long)f * 6 + nt6) * 16;
            v8f acc0 = {}, acc1 = {};
            v16bf bA  = load_b_glb(wsw + OFF_WOUT, base);
            v16bf a0A = load_a_lds(hC, HID, 0, 0);
            v16bf a1A = load_a_lds(hC, HID, 0, 16);
            #pragma unroll 2
            for (int kt = 0; kt < 16; kt += 2) {
                v16bf bB  = load_b_glb(wsw + OFF_WOUT, base + kt + 1);
                v16bf a0B = load_a_lds(hC, HID, (kt + 1) * 32, 0);
                v16bf a1B = load_a_lds(hC, HID, (kt + 1) * 32, 16);
                acc0 = WMMA_BF16(a0A, bA, acc0);
                acc1 = WMMA_BF16(a1A, bA, acc1);
                const int k2 = (kt + 2) & 15;
                bA  = load_b_glb(wsw + OFF_WOUT, base + k2);
                a0A = load_a_lds(hC, HID, k2 * 32, 0);
                a1A = load_a_lds(hC, HID, k2 * 32, 16);
                acc0 = WMMA_BF16(a0B, bB, acc0);
                acc1 = WMMA_BF16(a1B, bB, acc1);
            }
            const int col = nt6 * 16 + (lane & 15);
            const float bias = boutp[f * NPADF + col];
            #pragma unroll
            for (int r = 0; r < 8; ++r) {
                pw[(r + mhi) * NPADF + col]      = acc0[r] + bias;
                pw[(16 + r + mhi) * NPADF + col] = acc1[r] + bias;
            }
        }
        __syncthreads();

        {   // rational-quadratic spline for (row = lane, feature f)
            const float* pp = pw + lane * NPADF;
            const float x = pred[(long)(row0 + lane) * FEAT + f];

            // widths: softmax -> streaming cumsum + bin search
            float mw = -1e30f;
            for (int k = 0; k < NBINS; ++k) mw = fmaxf(mw, pp[k] * pscale);
            float sw = 0.f;
            for (int k = 0; k < NBINS; ++k) sw += __expf(pp[k] * pscale - mw);
            const float invw = c1 / sw;
            float e = 0.f, in_cw = 0.f, in_w = 1e-3f;
            int idx = 0;
            for (int k = 0; k < NBINS; ++k) {
                float wk = 1e-3f + __expf(pp[k] * pscale - mw) * invw;
                float en = (k == NBINS - 1) ? 1.0f : e + wk;
                if (x >= e) { idx = k; in_cw = e; in_w = en - e; }
                e = en;
            }
            // heights: same, capture at idx
            float mh = -1e30f;
            for (int k = 0; k < NBINS; ++k) mh = fmaxf(mh, pp[NBINS + k] * pscale);
            float sh = 0.f;
            for (int k = 0; k < NBINS; ++k) sh += __expf(pp[NBINS + k] * pscale - mh);
            const float invh = c1 / sh;
            float eh = 0.f, in_ch = 0.f, in_h = 1e-3f;
            for (int k = 0; k < NBINS; ++k) {
                float hk = 1e-3f + __expf(pp[NBINS + k] * pscale - mh) * invh;
                float en = (k == NBINS - 1) ? 1.0f : eh + hk;
                if (k == idx) { in_ch = eh; in_h = en - eh; }
                eh = en;
            }
            const float in_d   = 1e-3f + softplus_f(pp[2 * NBINS + idx]);
            const float in_dp1 = 1e-3f + softplus_f(pp[2 * NBINS + idx + 1]);
            const float th = (x - in_cw) / in_w;
            const float tt = th * (1.f - th);
            const float dl = in_h / in_w;
            const float num = in_h * (dl * th * th + in_d * tt);
            const float den = dl + (in_d + in_dp1 - 2.f * dl) * tt;
            float cdf = in_ch + num / den;
            cdf = (x == 1.0f) ? 1.0f : cdf;
            prod *= cdf;
        }
        __syncthreads();
    }

    parts[wave * ROWS + lane] = prod;
    __syncthreads();
    if (tid < ROWS) {
        float p = 1.0f;
        #pragma unroll
        for (int w = 0; w < 8; ++w) p *= parts[w * ROWS + tid];
        out[row0 + tid] = p;
    }
}

// ---------------------------------------------------------------------------
extern "C" void kernel_launch(void* const* d_in, const int* in_sizes, int n_in,
                              void* d_out, int out_size, void* d_ws, size_t ws_size,
                              hipStream_t stream) {
    const float* pred  = (const float*)d_in[0];
    const float* ctx   = (const float*)d_in[1];
    const float* W_in  = (const float*)d_in[2];
    const float* b_in  = (const float*)d_in[3];
    const float* Wc_in = (const float*)d_in[4];
    const float* bc_in = (const float*)d_in[5];
    const float* Wb1   = (const float*)d_in[6];
    const float* bb1   = (const float*)d_in[7];
    const float* Wb2   = (const float*)d_in[8];
    const float* bb2   = (const float*)d_in[9];
    const float* Wcb   = (const float*)d_in[10];
    const float* bcb   = (const float*)d_in[11];
    const float* W_out = (const float*)d_in[12];
    const float* b_out = (const float*)d_in[13];

    unsigned short* wsw = (unsigned short*)d_ws;
    float* boutp = (float*)((char*)d_ws + OFF_BOUT_BYTES);
    float* out   = (float*)d_out;

    const int B = in_sizes[0] / FEAT;

    prep_weights<<<128,  256, 0, stream>>>(W_in,  wsw + OFF_WIN,  64,  HID, HID, 1);
    prep_weights<<<1024, 256, 0, stream>>>(Wc_in, wsw + OFF_WCIN, HID, HID, HID, 0);
    for (int i = 0; i < 3; ++i) {
        prep_weights<<<1024, 256, 0, stream>>>(Wb1 + (long)i * 262144, wsw + OFF_WB1 + (long)i * 262144, HID, HID, HID, 2);
        prep_weights<<<1024, 256, 0, stream>>>(Wb2 + (long)i * 262144, wsw + OFF_WB2 + (long)i * 262144, HID, HID, HID, 2);
        prep_weights<<<1024, 256, 0, stream>>>(Wcb + (long)i * 262144, wsw + OFF_WCB + (long)i * 262144, HID, HID, HID, 0);
    }
    prep_weights<<<4096, 256, 0, stream>>>(W_out, wsw + OFF_WOUT, HID, FEAT * MULTP, NOUTP, 3);
    prep_bout<<<(NOUTP + 255) / 256, 256, 0, stream>>>(b_out, boutp);

    made_spline_kernel<<<B / ROWS, 256, 0, stream>>>(pred, ctx, b_in, bc_in,
                                                     bb1, bb2, bcb, wsw, boutp, out);
}
